// nonConsecutiveConvLayer2_56796647522675
// MI455X (gfx1250) — compile-verified, matrix operand order checked by reading, not run on hardware
//
#include <hip/hip_runtime.h>
#include <hip/hip_bf16.h>

// Problem constants (reference: B=64, L=256, D=1024, F=1024)
#define B_ 64
#define L_ 256
#define D_ 1024
#define F_ 1024
// Padded LDS row stride (bf16 elements): 1032*2 = 2064 B/row = 129*16
// -> ds_load_b128 stays 16B-aligned, and rows advance 4 banks each so the
// 16 rows a wave touches per b128 operand load hit distinct bank groups.
#define KP 1032

typedef __attribute__((ext_vector_type(16))) __bf16 v16bf;
typedef __attribute__((ext_vector_type(8)))  __bf16 v8bf;
typedef __attribute__((ext_vector_type(4)))  __bf16 v4bf;
typedef __attribute__((ext_vector_type(8)))  float  v8f;

// Build a v16bf WMMA operand from two 16-byte LDS loads (ds_load_b128 x2).
// Per ISA 16-bit A layout: lane holds K groups [0..7]+[16..23] (lanes 0-15)
// or [8..15]+[24..31] (lanes 16-31) -> +16-element offset between halves.
__device__ __forceinline__ v16bf ld_tile16(const __bf16* p) {
  v8bf lo = *(const v8bf*)(p);
  v8bf hi = *(const v8bf*)(p + 16);
  return __builtin_shufflevector(lo, hi, 0,1,2,3,4,5,6,7,8,9,10,11,12,13,14,15);
}

// 64 WMMAs: a_t = x_t @ W0tile, c_t = x_t @ W1tile (K=1024), then the
// max-scan update (old h1 feeds h2, matching the reference ordering).
__device__ __forceinline__ void compute_step(const __bf16* pA,
                                             const __bf16* pB0,
                                             const __bf16* pB1,
                                             v8f& h1, v8f& h2) {
  v8f a = {0.f,0.f,0.f,0.f,0.f,0.f,0.f,0.f};
  v8f c = {0.f,0.f,0.f,0.f,0.f,0.f,0.f,0.f};
  #pragma unroll
  for (int kb = 0; kb < D_; kb += 32) {
    v16bf A   = ld_tile16(pA  + kb);
    v16bf Bm0 = ld_tile16(pB0 + kb);
    v16bf Bm1 = ld_tile16(pB1 + kb);
    a = __builtin_amdgcn_wmma_f32_16x16x32_bf16(false, A, false, Bm0,
                                                (short)0, a, false, false);
    c = __builtin_amdgcn_wmma_f32_16x16x32_bf16(false, A, false, Bm1,
                                                (short)0, c, false, false);
  }
  #pragma unroll
  for (int r = 0; r < 8; ++r) {
    h2[r] = fmaxf(h2[r], h1[r] + c[r]);
    h1[r] = fmaxf(h1[r], a[r]);
  }
}

// Per-thread slice of an x[t] tile kept live in VGPRs across the compute
// phase (32 global_load_b128 issued early; s_wait_loadcnt lands ~500 WMMA
// cycles later -> global latency fully hidden). 128 VGPRs, fine at 1 wave/SIMD.
struct XRegs { float4 v[32]; };

__device__ __forceinline__ void load_x(const float4* __restrict__ src, XRegs& r) {
  #pragma unroll
  for (int i = 0; i < 32; ++i) r.v[i] = src[i];
}

__device__ __forceinline__ void store_x(const XRegs& r, __bf16* dst) {
  #pragma unroll
  for (int i = 0; i < 32; ++i) {
    float4 v = r.v[i];
    v4bf b4 = { (__bf16)v.x, (__bf16)v.y, (__bf16)v.z, (__bf16)v.w };
    *(v4bf*)(dst + i * 4) = b4;                    // ds_store_b64
  }
}

// One workgroup = 4 waves (128 threads), one 16x16 output tile per wave:
//   m-tile = blockIdx.x & 3 (B rows), n-tile = (blockIdx.x>>2)*4 + wave.
// LDS: per-wave transposed bf16 W0/W1 tiles (resident across all 256 steps)
//      + one shared 16x1024 bf16 x[t] tile restaged each step (290KB total).
__global__ void __launch_bounds__(128, 1)
fused_gemm_scan_tanh(const float* __restrict__ x,
                     const float* __restrict__ W0,
                     const float* __restrict__ W1,
                     const float* __restrict__ bias,
                     float* __restrict__ out) {
  extern __shared__ __bf16 smem[];

  const int tid   = threadIdx.x;
  const int lane  = tid & 31;
  const int wave  = tid >> 5;
  const int mtile = blockIdx.x & 3;   // B/16 = 4
  const int nblk  = blockIdx.x >> 2;  // F/(16*4) = 16
  const int nbase = (nblk * 4 + wave) * 16;

  __bf16* w0t = smem + (size_t)wave * (2 * 16 * KP);
  __bf16* w1t = w0t + 16 * KP;
  __bf16* xt  = smem + (size_t)8 * 16 * KP;   // 4 waves * 2 matrices = 8 tiles

  // ---- Stage W0/W1 n-tiles, transposed to [n][k], f32 -> bf16 (one time) ----
  {
    const int nn = lane & 15;          // column within tile
    const int kh = lane >> 4;          // two k-rows in flight per wave
    const float* p0 = W0 + nbase + nn;
    const float* p1 = W1 + nbase + nn;
    #pragma unroll 4
    for (int k = kh; k < D_; k += 2) {
      w0t[nn * KP + k] = (__bf16)p0[(size_t)k * F_];
      w1t[nn * KP + k] = (__bf16)p1[(size_t)k * F_];
    }
  }

  // Per-lane WMMA operand base pointers (A from shared x tile, B from W tiles).
  const int tcol  = lane & 15;
  const int khalf = (lane >> 4) << 3;              // 0 or 8
  const __bf16* pA  = xt  + tcol * KP + khalf;
  const __bf16* pB0 = w0t + tcol * KP + khalf;
  const __bf16* pB1 = w1t + tcol * KP + khalf;

  // Scan state: 16x16 f32 tiles in WMMA C/D register layout.
  v8f h1 = {0.f,0.f,0.f,0.f,0.f,0.f,0.f,0.f};
  v8f h2 = {0.f,0.f,0.f,0.f,0.f,0.f,0.f,0.f};

  // Cooperative x-staging geometry: 128 threads, 8 threads/row, 128 cols each.
  const int    row      = tid >> 3;                 // 0..15
  const int    kcol     = (tid & 7) * 128;
  const size_t xrowbase = (size_t)(mtile * 16 + row) * (size_t)(L_ * D_) + kcol;
  const float* xsrc     = x + xrowbase;
  __bf16*      xdst     = xt + row * KP + kcol;

  // ---- Software-pipelined t loop: loads for t+1 issue before compute(t) ----
  XRegs xr;
  load_x((const float4*)xsrc, xr);                  // t = 0
  __builtin_prefetch(xsrc + D_, 0, 3);
  store_x(xr, xdst);
  __syncthreads();                                  // xt(0) + W tiles ready

  for (int t = 0; t < L_ - 1; ++t) {
    load_x((const float4*)(xsrc + (size_t)(t + 1) * D_), xr);  // hide behind WMMAs
    if (t + 2 < L_) __builtin_prefetch(xsrc + (size_t)(t + 2) * D_, 0, 3);

    compute_step(pA, pB0, pB1, h1, h2);             // consumes xt(t)

    __syncthreads();                                // all waves done reading xt
    store_x(xr, xdst);                              // convert + ds_store xt(t+1)
    __syncthreads();                                // xt(t+1) visible
  }
  compute_step(pA, pB0, pB1, h1, h2);               // t = L_-1

  // ---- Epilogue: out[b, f] = tanh(h2 + bias), C/D VGPR layout ----
  const int   n    = lane & 15;
  const int   mofs = (lane >> 4) * 8;  // VGPR r: lanes 0-15 -> M=r, 16-31 -> M=r+8
  const float bb   = bias[nbase + n];
  #pragma unroll
  for (int r = 0; r < 8; ++r) {
    const int brow = mtile * 16 + mofs + r;
    out[(size_t)brow * F_ + nbase + n] = tanhf(h2[r] + bb);
  }
}

extern "C" void kernel_launch(void* const* d_in, const int* in_sizes, int n_in,
                              void* d_out, int out_size, void* d_ws, size_t ws_size,
                              hipStream_t stream) {
  const float* x    = (const float*)d_in[0];   // [B, L, D]
  const float* W0   = (const float*)d_in[1];   // [D, F]
  const float* W1   = (const float*)d_in[2];   // [D, F]
  const float* bias = (const float*)d_in[3];   // [F]
  float* out = (float*)d_out;                  // [B, F]

  (void)in_sizes; (void)n_in; (void)out_size; (void)d_ws; (void)ws_size;

  // 8 W tiles + 1 x tile of 16 rows x KP bf16 = 144*1032*2 = 297216 B (<320KB)
  const size_t smem_bytes = (size_t)(8 * 16 + 16) * KP * sizeof(__bf16);
  dim3 grid(64);    // 4 m-tiles * 16 n-blocks
  dim3 block(128);  // 4 waves, one 16x16 output tile each
  fused_gemm_scan_tanh<<<grid, block, smem_bytes, stream>>>(x, W0, W1, bias, out);
}
